// GaussianTrans_67156108640826
// MI455X (gfx1250) — matrix operand order repaired and checked by reading, not compiled, not run on hardware
//
#include <hip/hip_runtime.h>

// MI455X / gfx1250, wave32. Fused gaussian-bias softmax + WMMA f16 matmul,
// TDM (tensor_load_to_lds) for the attention tile staging.

typedef __attribute__((ext_vector_type(16))) _Float16 v16h;
typedef __attribute__((ext_vector_type(8)))  _Float16 v8h;
typedef __attribute__((ext_vector_type(8)))  float    v8f;
typedef __attribute__((ext_vector_type(4)))  float    f32x4;
typedef __attribute__((ext_vector_type(4)))  unsigned int u32x4;
typedef __attribute__((ext_vector_type(8)))  int      i32x8;
typedef __attribute__((ext_vector_type(4)))  int      i32x4;

#define HD 128          // H == W == 128
#define CD 256          // channel dim
#define AS_STRIDE 136   // f16 halves, 16B-aligned rows
#define BS_STRIDE 136
#define STAGE_STRIDE 132 // f32 words; matches TDM pad (128 dw + 4 dw pad)

#define BS_BYTES (CD * BS_STRIDE * 2)   /* 69632; also covers f32 stage (67584) */
#define AS_BYTES (HD * AS_STRIDE * 2)   /* 34816 */
#define RS_BYTES (HD * 4)               /* 512   */

union HF16 { v16h v; v8h h[2]; };

template<bool ACCUM>
__global__ __launch_bounds__(256)
void gtrans_kernel(const float* __restrict__ atten_all,
                   const float* __restrict__ value,
                   const float* __restrict__ shiftp,
                   const float* __restrict__ biasp,
                   float* __restrict__ out)
{
    __shared__ __align__(16) char smem[BS_BYTES + AS_BYTES + RS_BYTES];
    float*    stage    = (float*)smem;                      // 128 x 132 f32 (phase 1-2)
    _Float16* Bs       = (_Float16*)smem;                   // 256 x 136 f16 (phase 3+), aliases stage
    _Float16* As       = (_Float16*)(smem + BS_BYTES);      // 128 x 136 f16
    float*    rowscale = (float*)(smem + BS_BYTES + AS_BYTES);

    const int t = threadIdx.x;
    const int b = blockIdx.x >> 7;
    const int i = blockIdx.x & 127;

    const float sh = shiftp[0];
    const float bi = biasp[0];

    // attention tile [128][128] is contiguous for both tensors
    const float* atten = atten_all + (size_t)blockIdx.x * (HD * HD);

    const float* vbase;
    float*       obase;
    size_t       vstride, ostride;
    if (ACCUM) {   // kernel 2: fixed column c=i; V rows strided by W*C
        vbase   = value + ((size_t)b * (HD * HD) + (size_t)i) * CD;
        vstride = (size_t)HD * CD;
        obase   = out   + ((size_t)b * (HD * HD) + (size_t)i) * CD;
        ostride = (size_t)HD * CD;
    } else {       // kernel 1: fixed row r=i; V tile contiguous
        vbase   = value + (size_t)(b * HD + i) * HD * CD;
        vstride = CD;
        obase   = out   + (size_t)(b * HD + i) * HD * CD;
        ostride = CD;
    }

    // ---------- Phase 1: TDM DMA of the 64KB attention tile into padded LDS ----------
    if (t < 32) {  // wave 0 issues the tensor DMA (EXEC ignored by TDM)
        unsigned long long ga = (unsigned long long)(uintptr_t)atten;
        unsigned int ldsoff   = (unsigned int)(uintptr_t)(void*)smem;   // LDS byte offset

        u32x4 g0;
        g0[0] = 1u;                                 // count=1, no flags
        g0[1] = ldsoff;                             // lds_addr
        g0[2] = (unsigned int)ga;                   // global_addr[31:0]
        g0[3] = (unsigned int)((ga >> 32) & 0x01FFFFFFu) | (2u << 30); // [56:32] | type=2

        i32x8 g1;
        g1[0] = (2 << 16)        // data_size = 4B
              | (1 << 20)        // pad_enable
              | (6 << 22)        // pad_interval: 128 DWORDs
              | (3 << 25);       // pad_amount:   4 DWORDs  -> row stride 132 f32
        g1[1] = (HD << 16);      // tensor_dim0[15:0] in [63:48]
        g1[2] = (HD << 16);      // tensor_dim1[15:0] in [95:80]
        g1[3] = (HD << 16);      // tile_dim0 in [127:112]
        g1[4] = HD;              // tile_dim1 in [143:128]; tile_dim2 = 0
        g1[5] = HD;              // tensor_dim0_stride[31:0]
        g1[6] = 0;
        g1[7] = 0;

        i32x4 gz = (i32x4)0;     // groups 2/3 unused (2D tensor)

        asm volatile("tensor_load_to_lds %0, %1, %2, %3"
                     :: "s"(g0), "s"(g1), "s"(gz), "s"(gz)
                     : "memory");
        __builtin_amdgcn_s_wait_tensorcnt(0);
    }
    __syncthreads();

    // ---------- Phase 2: row softmax with gaussian bias; unnormalized exp -> f16 ----------
    {
        int row = t >> 1;
        int kb  = (t & 1) * 64;
        const float* srow = stage + row * STAGE_STRIDE;
        float m = -3.4e38f;
        #pragma unroll 4
        for (int k = 0; k < 64; ++k) {
            float dk  = (float)(kb + k - row);
            float val = srow[kb + k] - fmaf(sh, dk * dk, bi);
            m = fmaxf(m, val);
        }
        m = fmaxf(m, __shfl_xor(m, 1, 32));
        float s = 0.f;
        _Float16* arow = As + row * AS_STRIDE;
        #pragma unroll 4
        for (int k = 0; k < 64; k += 2) {
            float dk0 = (float)(kb + k - row);
            float dk1 = (float)(kb + k + 1 - row);
            float v0  = srow[kb + k]     - fmaf(sh, dk0 * dk0, bi);
            float v1  = srow[kb + k + 1] - fmaf(sh, dk1 * dk1, bi);
            float e0  = __expf(v0 - m);
            float e1  = __expf(v1 - m);
            union { _Float16 h[2]; unsigned int u; } pk;
            pk.h[0] = (_Float16)e0;
            pk.h[1] = (_Float16)e1;
            *(unsigned int*)(arow + kb + k) = pk.u;   // packed b32 store
            s += e0 + e1;
        }
        s += __shfl_xor(s, 1, 32);
        if ((t & 1) == 0) rowscale[row] = 1.0f / s;
    }
    __syncthreads();

    // ---------- Phase 3: load V tile (2 rows/thread), transpose-convert to Bs[n][k] ----------
    #pragma unroll
    for (int j = 0; j < 16; ++j) {
        int idx = t + 256 * j;            // 0..4095
        int k   = (idx >> 6) * 2;         // even k; two rows per thread
        int c4  = idx & 63;               // 64 float4 per row of 256
        const float* p0 = vbase + (size_t)k * vstride + c4 * 4;
        f32x4 a, c;
        if (ACCUM) {                      // kernel 2: V touched once -> NT
            a = __builtin_nontemporal_load((const f32x4*)p0);
            c = __builtin_nontemporal_load((const f32x4*)(p0 + vstride));
        } else {
            a = *(const f32x4*)p0;
            c = *(const f32x4*)(p0 + vstride);
        }
        int n = c4 * 4;
        union { _Float16 h[2]; unsigned int u; } pk;
        #pragma unroll
        for (int q = 0; q < 4; ++q) {
            pk.h[0] = (_Float16)a[q];
            pk.h[1] = (_Float16)c[q];
            *(unsigned int*)(Bs + (n + q) * BS_STRIDE + k) = pk.u;
        }
    }
    __syncthreads();

    // ---------- Phase 4: WMMA matmul. Wave w owns rows [16w,16w+16) x all 256 cols ----------
    const int lane = t & 31;
    const int w    = t >> 5;
    const int m0   = w * 16;
    const int lsel = (lane < 16) ? 0 : 1;

    v8f acc[16];
    #pragma unroll
    for (int nt = 0; nt < 16; ++nt) acc[nt] = 0.f;

    const _Float16* Arow = As + (m0 + (lane & 15)) * AS_STRIDE;
    const int kA = lsel ? 8 : 0;     // A: lanes 0-15 K{0-7,16-23}; lanes 16-31 K{8-15,24-31}
    const int kB = lsel ? 8 : 0;     // B: lanes 0-15 K{0-7,16-23}; lanes 16-31 K{8-15,24-31}

    #pragma unroll
    for (int ks = 0; ks < HD; ks += 32) {
        HF16 af;
        af.h[0] = *(const v8h*)(Arow + ks + kA);
        af.h[1] = *(const v8h*)(Arow + ks + kA + 16);
        #pragma unroll
        for (int nt = 0; nt < 16; ++nt) {
            const _Float16* bp = Bs + (nt * 16 + (lane & 15)) * BS_STRIDE + ks + kB;
            HF16 bf;
            bf.h[0] = *(const v8h*)(bp);
            bf.h[1] = *(const v8h*)(bp + 16);
            acc[nt] = __builtin_amdgcn_wmma_f32_16x16x32_f16(
                false, af.v, false, bf.v, (short)0, acc[nt], false, false);
        }
    }

    // ---------- Phase 5: scale by 1/rowsum (softmax normalization) and store ----------
    float scale[8];
    #pragma unroll
    for (int q = 0; q < 8; ++q)
        scale[q] = rowscale[m0 + q + (lsel ? 8 : 0)];

    #pragma unroll
    for (int nt = 0; nt < 16; ++nt) {
        int n = nt * 16 + (lane & 15);
        #pragma unroll
        for (int q = 0; q < 8; ++q) {
            int m = m0 + q + (lsel ? 8 : 0);
            float vv = acc[nt][q] * scale[q];
            float* p = obase + (size_t)m * ostride + n;
            if (ACCUM) {
                // disjoint ownership per workgroup -> race-free RMW; result is final -> NT
                float old = __builtin_nontemporal_load(p);
                __builtin_nontemporal_store(old + vv, p);
            } else {
                *p = vv;   // re-read by kernel 2 -> keep RT for L2 residency
            }
        }
    }
}

extern "C" void kernel_launch(void* const* d_in, const int* in_sizes, int n_in,
                              void* d_out, int out_size, void* d_ws, size_t ws_size,
                              hipStream_t stream) {
    // setup_inputs order: x, atten_x_full, atten_y_full, value_full, shift, bias
    const float* atten_x = (const float*)d_in[1];
    const float* atten_y = (const float*)d_in[2];
    const float* value   = (const float*)d_in[3];
    const float* shift   = (const float*)d_in[4];
    const float* bias    = (const float*)d_in[5];
    float* out = (float*)d_out;

    dim3 grid(8 * HD);   // one workgroup per (b, r) / (b, c)
    dim3 block(256);     // 8 wave32

    gtrans_kernel<false><<<grid, block, 0, stream>>>(atten_x, value, shift, bias, out);
    gtrans_kernel<true ><<<grid, block, 0, stream>>>(atten_y, value, shift, bias, out);
}
